// GATLayer_17025250361522
// MI455X (gfx1250) — matrix operand order repaired and checked by reading, not compiled, not run on hardware
//
#include <hip/hip_runtime.h>
#include <hip/hip_bf16.h>
#include <stdint.h>

#define N_NODES 8192
#define F_IN    256
#define H_HEADS 4
#define D_HEAD  64
#define HD      256          // H_HEADS * D_HEAD
#define JC      32           // j-chunk (WMMA K)
#define LEAKY   0.2f

// LDS row pitches (elements) chosen for 16B alignment + conflict-free banks
#define HB_PITCH  40         // 32 bf16 data + 8 pad  = 80 B  (20 DW stride)
#define ADJ_PITCH 36         // 32 i32  data + 4 pad  = 144 B (36 DW stride)

typedef __bf16    v16bf __attribute__((ext_vector_type(16)));
typedef __bf16    v8bf  __attribute__((ext_vector_type(8)));
typedef float     v8f   __attribute__((ext_vector_type(8)));
typedef uint32_t  v4u   __attribute__((ext_vector_type(4)));
typedef int       v8i   __attribute__((ext_vector_type(8)));
typedef int       v4i   __attribute__((ext_vector_type(4)));

union BfFrag { v16bf v; v8bf h[2]; __bf16 x[16]; };
union I16v   { int4 v4[4];  int   s[16]; };
union F16v   { float4 v4[4]; float s[16]; };

#if defined(__has_builtin)
#if __has_builtin(__builtin_amdgcn_tensor_load_to_lds) && __has_builtin(__builtin_amdgcn_s_wait_tensorcnt)
#define USE_TDM 1
#endif
#endif

#ifdef USE_TDM
// 2D tensor-tile DMA: global -> LDS, optional LDS row padding.
// ds_code: 0=1B 1=2B 2=4B. tile0 elems contiguous, tile1 rows, stride0 in elems.
// pad_int: 0=off else code n -> pad after 2^(n+1) DWORDs; pad_amt code -> (code+1) DWORDs.
__device__ inline void tdm_2d(uint32_t lds_off, uint64_t gaddr, uint32_t ds_code,
                              uint32_t tile0, uint32_t tile1,
                              uint64_t tdim0, uint64_t tdim1, uint64_t stride0,
                              uint32_t pad_en, uint32_t pad_int, uint32_t pad_amt) {
  v4u g0;
  g0[0] = 1u;                                   // count=1, user mode
  g0[1] = lds_off;                              // lds_addr  [63:32]
  g0[2] = (uint32_t)gaddr;                      // global_addr[95:64]
  g0[3] = (uint32_t)((gaddr >> 32) & 0x01FFFFFFull) | (2u << 30); // addr[56:32]|type=2
  v8i g1;
  g1[0] = (int)((ds_code << 16) | (pad_en << 20) | (pad_int << 22) | (pad_amt << 25));
  g1[1] = (int)((tdim0 & 0xFFFFull) << 16);                   // tensor_dim0[15:0]
  g1[2] = (int)(((tdim0 >> 16) & 0xFFFFull) | ((tdim1 & 0xFFFFull) << 16));
  g1[3] = (int)(((tdim1 >> 16) & 0xFFFFull) | ((uint64_t)tile0 << 16)); // tile_dim0
  g1[4] = (int)(tile1 & 0xFFFFu);                             // tile_dim1 (tile_dim2=0)
  g1[5] = (int)(stride0 & 0xFFFFFFFFull);                     // tensor_dim0_stride lo
  g1[6] = (int)((stride0 >> 32) & 0xFFFFull);                 // stride hi (dim1_stride=0)
  g1[7] = 0;
  v4i z4 = {0, 0, 0, 0};
  v8i z8 = {0, 0, 0, 0, 0, 0, 0, 0};
  __builtin_amdgcn_tensor_load_to_lds(g0, g1, z4, z4, z8, 0);
}
#endif

// ---------------- Kernel A: h = x @ W (bf16 WMMA, f32 accum) ---------------
// Writes h (row-major f32) and hbfT (bf16, TRANSPOSED: [col][row]).
__global__ __launch_bounds__(128) void gemm_proj(const float* __restrict__ x,
                                                 const float* __restrict__ W,
                                                 float* __restrict__ h,
                                                 __bf16* __restrict__ hbfT) {
  int wave = threadIdx.x >> 5;          // 4 waves -> 4 column groups of 64 (one head)
  int lane = threadIdx.x & 31;
  int lrow = lane & 15, lhi = lane >> 4;
  int i0 = blockIdx.x * 16;
  int c0 = wave * 64;

  v8f acc[4] = {};
  for (int kc = 0; kc < F_IN; kc += 32) {
    v16bf a;
#pragma unroll
    for (int t = 0; t < 16; ++t) {      // A 16x32: K = (t/8)*16 + lhi*8 + t%8
      int k = kc + ((t >> 3) << 4) + (lhi << 3) + (t & 7);
      a[t] = (__bf16)x[(uint64_t)(i0 + lrow) * F_IN + k];
    }
#pragma unroll
    for (int nt = 0; nt < 4; ++nt) {
      v16bf b;
#pragma unroll
      for (int t = 0; t < 16; ++t) {    // B 32x16: K = (t/8)*16 + lhi*8 + t%8, N = lrow
        int k = kc + ((t >> 3) << 4) + (lhi << 3) + (t & 7);
        b[t] = (__bf16)W[(uint64_t)k * HD + c0 + nt * 16 + lrow];
      }
      acc[nt] = __builtin_amdgcn_wmma_f32_16x16x32_bf16(
          false, a, false, b, (short)0, acc[nt], false, false);
    }
  }
#pragma unroll
  for (int nt = 0; nt < 4; ++nt)
#pragma unroll
    for (int r = 0; r < 8; ++r) {
      int m = r + (lhi << 3);
      int col = c0 + nt * 16 + lrow;
      float v = acc[nt][r];
      h[(uint64_t)(i0 + m) * HD + col] = v;
      hbfT[(uint64_t)col * N_NODES + i0 + m] = (__bf16)v;
    }
}

// ---------------- Kernel S: score_l (n-major) / score_r (head-major) -------
__global__ void scores_k(const float* __restrict__ h, const float* __restrict__ a_l,
                         const float* __restrict__ a_r, float* __restrict__ sl,
                         float* __restrict__ srT) {
  int t = blockIdx.x * blockDim.x + threadIdx.x;   // n*H + head
  if (t >= N_NODES * H_HEADS) return;
  int n = t >> 2, hh = t & 3;
  const float* hp = h + (uint64_t)n * HD + hh * D_HEAD;
  float accl = 0.f, accr = 0.f;
#pragma unroll 8
  for (int d = 0; d < D_HEAD; ++d) {
    float v = hp[d];
    accl += v * a_l[hh * D_HEAD + d];
    accr += v * a_r[hh * D_HEAD + d];
  }
  sl[t] = accl;
  srT[hh * N_NODES + n] = accr;
}

// ---------------- Kernel M: per-head max of score_r ------------------------
__global__ void srmax_k(const float* __restrict__ srT, float* __restrict__ srmax) {
  __shared__ float sm[256];
  int tid = threadIdx.x;
  int hh = tid & 3;
  float m = -1e30f;
  for (int n = tid >> 2; n < N_NODES; n += 64)
    m = fmaxf(m, srT[hh * N_NODES + n]);
  sm[tid] = m;
  __syncthreads();
  for (int s = 128; s >= 4; s >>= 1) {
    if (tid < s) sm[tid] = fmaxf(sm[tid], sm[tid + s]);
    __syncthreads();
  }
  if (tid < 4) srmax[tid] = sm[tid];
}

// ---------------- Kernel C: masked softmax + aggregation -------------------
__global__ __launch_bounds__(128) void gat_aggregate(
    const int* __restrict__ adj, const __bf16* __restrict__ hbfT,
    const float* __restrict__ sl, const float* __restrict__ srT,
    const float* __restrict__ srmax, float* __restrict__ out) {
  __shared__ int    adj_s[2][16][ADJ_PITCH];      // 4.5 KB  (padded rows)
  __shared__ __bf16 hbf_s[2][HD][HB_PITCH];       // 40 KB   (padded rows, [col][j])
  __shared__ float  srT_s[2][H_HEADS][JC];        // 1 KB
  __shared__ float  rowsum_s[H_HEADS][16];

  int wave = threadIdx.x >> 5;          // wave == head
  int lane = threadIdx.x & 31;
  int lrow = lane & 15, lhi = lane >> 4;
  int i0 = blockIdx.x * 16;
  int hh = wave;

  float sl_lane = sl[(i0 + lrow) * H_HEADS + hh];
  float cmax = sl_lane + srmax[hh];               // leaky monotone -> row upper bound
  cmax = cmax > 0.f ? cmax : LEAKY * cmax;

  const uint64_t adj_base = (uint64_t)(uintptr_t)adj;
  const uint64_t hbf_base = (uint64_t)(uintptr_t)hbfT;
  const uint64_t srT_base = (uint64_t)(uintptr_t)srT;

  v8f acc[4] = {};
  float rs = 0.f;

  auto issue = [&](int j0, int buf) {
#ifdef USE_TDM
    if (wave == 0) {
      // adj tile: 16 rows(i) x 32 cols(j), pad rows 32DW -> +4DW (pitch 36 DW)
      tdm_2d((uint32_t)(uintptr_t)&adj_s[buf][0][0],
             adj_base + ((uint64_t)i0 * N_NODES + j0) * 4ull,
             2u, JC, 16, N_NODES, N_NODES, N_NODES, 1u, 4u, 3u);
      // hT tile: 256 rows(col) x 32 j, pad rows 16DW -> +4DW (pitch 20 DW)
      tdm_2d((uint32_t)(uintptr_t)&hbf_s[buf][0][0],
             hbf_base + (uint64_t)j0 * 2ull,
             1u, JC, HD, N_NODES, HD, N_NODES, 1u, 3u, 3u);
      // srT tile: 4 rows(head) x 32 j, no pad
      tdm_2d((uint32_t)(uintptr_t)&srT_s[buf][0][0],
             srT_base + (uint64_t)j0 * 4ull,
             2u, JC, H_HEADS, N_NODES, H_HEADS, N_NODES, 0u, 0u, 0u);
    }
#else
    for (int idx = threadIdx.x; idx < 16 * JC; idx += blockDim.x)
      adj_s[buf][idx / JC][idx % JC] =
          adj[(uint64_t)(i0 + idx / JC) * N_NODES + j0 + idx % JC];
    for (int idx = threadIdx.x; idx < HD * JC; idx += blockDim.x)
      hbf_s[buf][idx / JC][idx % JC] =
          hbfT[(uint64_t)(idx / JC) * N_NODES + j0 + idx % JC];
    for (int idx = threadIdx.x; idx < H_HEADS * JC; idx += blockDim.x)
      srT_s[buf][idx / JC][idx % JC] = srT[(uint64_t)(idx / JC) * N_NODES + j0 + idx % JC];
#endif
  };

  const int nch = N_NODES / JC;
  issue(0, 0);
  for (int k = 0; k < nch; ++k) {
    int buf = k & 1;
    if (k + 1 < nch) {
      issue((k + 1) * JC, buf ^ 1);
#ifdef USE_TDM
      if (wave == 0) __builtin_amdgcn_s_wait_tensorcnt(3);  // chunk k arrived
#endif
    } else {
#ifdef USE_TDM
      if (wave == 0) __builtin_amdgcn_s_wait_tensorcnt(0);  // last chunk arrived
#endif
    }
    __syncthreads();

    int j0 = k * JC;
    // ---- A operands via wide LDS loads ----
    I16v aj;                       // adj row lrow, runs [lhi*8..+7] and [16+lhi*8..+7]
    aj.v4[0] = *(const int4*)&adj_s[buf][lrow][(lhi << 3)];
    aj.v4[1] = *(const int4*)&adj_s[buf][lrow][(lhi << 3) + 4];
    aj.v4[2] = *(const int4*)&adj_s[buf][lrow][16 + (lhi << 3)];
    aj.v4[3] = *(const int4*)&adj_s[buf][lrow][16 + (lhi << 3) + 4];
    F16v sv;                       // score_r for same j runs (broadcast per half-wave)
    sv.v4[0] = *(const float4*)&srT_s[buf][hh][(lhi << 3)];
    sv.v4[1] = *(const float4*)&srT_s[buf][hh][(lhi << 3) + 4];
    sv.v4[2] = *(const float4*)&srT_s[buf][hh][16 + (lhi << 3)];
    sv.v4[3] = *(const float4*)&srT_s[buf][hh][16 + (lhi << 3) + 4];

    // A fragment: P[i, jl] = keep * exp(leaky(sl_i + sr_j) - cmax_i)
    BfFrag aP;
#pragma unroll
    for (int t = 0; t < 16; ++t) {
      int jl = ((t >> 3) << 4) + (lhi << 3) + (t & 7);
      int keep = aj.s[t] | ((i0 + lrow) == (j0 + jl));
      float e = 0.f;
      if (keep) {
        float z = sl_lane + sv.s[t];
        z = z > 0.f ? z : LEAKY * z;
        e = __expf(z - cmax);
      }
      rs += e;
      aP.x[t] = (__bf16)e;
    }
    // ---- B fragments: runs [lhi*8..+7] and [16+lhi*8..+7] per lane (two ds_load_b128) ----
#pragma unroll
    for (int nt = 0; nt < 4; ++nt) {
      BfFrag b;
      const __bf16* rowp = &hbf_s[buf][hh * D_HEAD + nt * 16 + lrow][(lhi << 3)];
      b.h[0] = *(const v8bf*)rowp;
      b.h[1] = *(const v8bf*)(rowp + 16);
      acc[nt] = __builtin_amdgcn_wmma_f32_16x16x32_bf16(
          false, aP.v, false, b.v, (short)0, acc[nt], false, false);
    }
    __syncthreads();   // protect buf before it is refilled next iteration
  }

  // combine row sums across the two lane halves (lane L and L+16 share row)
  rs += __shfl_xor(rs, 16, 32);
  rowsum_s[hh][lrow] = rs;
  __syncthreads();

#pragma unroll
  for (int nt = 0; nt < 4; ++nt)
#pragma unroll
    for (int r = 0; r < 8; ++r) {
      int m = r + (lhi << 3);
      out[(uint64_t)(i0 + m) * HD + hh * D_HEAD + nt * 16 + lrow] =
          acc[nt][r] / rowsum_s[hh][m];
    }
}

// ---------------------------------------------------------------------------
extern "C" void kernel_launch(void* const* d_in, const int* in_sizes, int n_in,
                              void* d_out, int out_size, void* d_ws, size_t ws_size,
                              hipStream_t stream) {
  const float* x   = (const float*)d_in[0];
  const int*   adj = (const int*)d_in[1];
  const float* W   = (const float*)d_in[2];
  const float* a_l = (const float*)d_in[3];
  const float* a_r = (const float*)d_in[4];
  float* out = (float*)d_out;

  char* ws = (char*)d_ws;
  float*  h    = (float*)ws;                                       // 8 MB
  __bf16* hbfT = (__bf16*)(ws + (size_t)N_NODES * HD * 4);         // 4 MB
  float*  sl   = (float*)(ws + (size_t)N_NODES * HD * 6);
  float*  srT  = sl + N_NODES * H_HEADS;
  float*  srm  = srT + N_NODES * H_HEADS;

  gemm_proj<<<N_NODES / 16, 128, 0, stream>>>(x, W, h, hbfT);
  scores_k<<<(N_NODES * H_HEADS + 255) / 256, 256, 0, stream>>>(h, a_l, a_r, sl, srT);
  srmax_k<<<1, 256, 0, stream>>>(srT, srm);
  gat_aggregate<<<N_NODES / 16, 128, 0, stream>>>(adj, hbfT, sl, srT, srm, out);
}
